// WindowAttention_19121194402483
// MI455X (gfx1250) — compile-verified, matrix operand order checked by reading, not run on hardware
//
#include <hip/hip_runtime.h>
#include <hip/hip_bf16.h>

#define NW_TOTAL 4096
#define T 49
#define TP 64
#define C 128
#define H 4
#define HD 32
#define NWPI 64

typedef __attribute__((ext_vector_type(8)))  unsigned short u16x8;
typedef __attribute__((ext_vector_type(16))) __bf16         v16bf;
typedef __attribute__((ext_vector_type(8)))  float          f32x8;

// LDS row strides (16-bit / 32-bit units): 16B-aligned rows for b128 frag
// loads AND bank-conflict-free strided row access.
#define RS_X   136   // x / out tile, bf16, 64 rows   (272 B/row)
#define RS_QK  264   // q|k tile,    bf16, 64 rows   (528 B/row)
#define RS_VT  72    // v transposed bf16, 128 rows  (144 B/row)
#define RS_S   68    // scores f32,  4*64 rows       (272 B/row)

__device__ __forceinline__ unsigned short f2bf(float f) {
    unsigned int u = __builtin_bit_cast(unsigned int, f);
    u += 0x7FFFu + ((u >> 16) & 1u);          // round-to-nearest-even
    return (unsigned short)(u >> 16);
}

// Packed f32x2 -> bf16x2. Prefer the builtin; else emit v_cvt_pk_bf16_f32.
#if __has_builtin(__builtin_amdgcn_cvt_pk_bf16_f32)
__device__ __forceinline__ unsigned int f2bf2(float a, float b) {
    auto r = __builtin_amdgcn_cvt_pk_bf16_f32(a, b);
    return __builtin_bit_cast(unsigned int, r);
}
#else
__device__ __forceinline__ unsigned int f2bf2(float a, float b) {
    unsigned int r;
    asm("v_cvt_pk_bf16_f32 %0, %1, %2" : "=v"(r) : "v"(a), "v"(b));
    return r;
}
#endif

// 16-bit WMMA A/B fragment: 8 contiguous K values + 8 more at +16 elements.
__device__ __forceinline__ v16bf ldfrag(const unsigned short* p) {
    union { u16x8 h[2]; v16bf v; } f;
    f.h[0] = *(const u16x8*)(p);
    f.h[1] = *(const u16x8*)(p + 16);
    return f.v;
}

__global__ void wattn_prep_weights(const float* __restrict__ Wqkv,
                                   const float* __restrict__ Wproj,
                                   unsigned short* __restrict__ WqkvT,
                                   unsigned short* __restrict__ WprojT) {
    int i = blockIdx.x * blockDim.x + threadIdx.x;   // 65536 threads
    if (i < 3 * C * C) {                 // WqkvT[n][k] = Wqkv[k][n]
        int n = i >> 7, k = i & 127;
        WqkvT[i] = f2bf(Wqkv[k * (3 * C) + n]);
    } else {                             // WprojT[n][k] = Wproj[k][n]
        int j = i - 3 * C * C;
        int n = j >> 7, k = j & 127;
        WprojT[j] = f2bf(Wproj[k * C + n]);
    }
}

// Pad mask into [NWPI][TP][RS_S] f32: -1e30 outside the valid 49x49 block so
// softmax pad columns collapse to exp() == 0 with no extra masking work.
__global__ void wattn_prep_mask(const float* __restrict__ mask,
                                float* __restrict__ maskP) {
    int i = blockIdx.x * blockDim.x + threadIdx.x;   // NWPI*TP*RS_S threads
    int k  = i % RS_S;
    int r  = (i / RS_S) & 63;
    int mw = i / (RS_S * TP);
    float v = -1e30f;
    if (r < T && k < T) v = mask[(mw * T + r) * T + k];
    maskP[i] = v;
}

__global__ void __launch_bounds__(256, 2)
wattn_main(const float* __restrict__ x,
           const float* __restrict__ maskP,
           const float* __restrict__ bqkv,
           const float* __restrict__ bproj,
           const unsigned short* __restrict__ WqkvT,
           const unsigned short* __restrict__ WprojT,
           float* __restrict__ out) {
    __shared__ __align__(16) unsigned short sX[TP * RS_X];     // x bf16; reused as attn-out tile
    __shared__ __align__(16) unsigned short sQK[TP * RS_QK];   // cols 0..127 = q*scale, 128..255 = k
    __shared__ __align__(16) unsigned short sVT[C * RS_VT];    // v^T: row = c (=h*32+d), col = token
    __shared__ __align__(16) float          sS[H * TP * RS_S]; // scores f32; bf16 probs in place

    const int tid  = threadIdx.x;
    const int wave = tid >> 5;
    const int lane = tid & 31;
    const int lh   = lane & 15;
    const int kb   = (lane >> 4) << 3;    // lane half selects K chunk 0..7 vs 8..15

    const int w = blockIdx.x;
    const float* xw = x + (size_t)w * T * C;

    // ---- Phase A: stage x as bf16 into LDS (float4 loads, uint2 stores) ----
    for (int i = tid; i < TP * C / 4; i += 256) {
        int r = i >> 5, c4 = (i & 31) << 2;      // 32 float4 per row
        float4 v = make_float4(0.f, 0.f, 0.f, 0.f);
        if (r < T) v = *(const float4*)(xw + r * C + c4);
        uint2 p;
        p.x = f2bf2(v.x, v.y);
        p.y = f2bf2(v.z, v.w);
        *(uint2*)&sX[r * RS_X + c4] = p;
    }
    __syncthreads();

    // ---- Phase B1: q|k = x @ Wqkv[:, 0:256] + b  (4x16 tiles, K=128) ----
    //      1/sqrt(hd) folded into the q store.
    for (int t0 = wave; t0 < 64; t0 += 8) {
        const int t  = __builtin_amdgcn_readfirstlane(t0);
        const int m0 = (t >> 4) << 4, n0 = (t & 15) << 4;
        f32x8 acc = {};
        const unsigned short* arow = sX + (m0 + lh) * RS_X;
        const unsigned short* brow = WqkvT + (n0 + lh) * C;
        for (int kk = 0; kk < 4; ++kk) {
            v16bf a = ldfrag(arow + kk * 32 + kb);
            v16bf b = ldfrag(brow + kk * 32 + kb);
            acc = __builtin_amdgcn_wmma_f32_16x16x32_bf16(false, a, false, b,
                                                          (short)0, acc, false, false);
        }
        const float sc   = (n0 < C) ? 0.17677669529663687f : 1.0f;  // scale q only
        const float bias = bqkv[n0 + lh];
        const int n = n0 + lh;
        for (int r = 0; r < 8; r += 2) {
            unsigned int p = f2bf2((acc[r] + bias) * sc, (acc[r + 1] + bias) * sc);
            int m = m0 + r + kb;
            sQK[m * RS_QK + n]       = (unsigned short)p;
            sQK[(m + 1) * RS_QK + n] = (unsigned short)(p >> 16);
        }
    }

    // ---- Phase B2: v = x @ Wqkv[:, 256:384] + b, stored transposed ----
    for (int t0 = wave; t0 < 32; t0 += 8) {
        const int t  = __builtin_amdgcn_readfirstlane(t0);
        const int m0 = (t >> 3) << 4, n0 = (t & 7) << 4;   // n0 within v's 128 cols
        f32x8 acc = {};
        const unsigned short* arow = sX + (m0 + lh) * RS_X;
        const unsigned short* brow = WqkvT + (2 * C + n0 + lh) * C;
        for (int kk = 0; kk < 4; ++kk) {
            v16bf a = ldfrag(arow + kk * 32 + kb);
            v16bf b = ldfrag(brow + kk * 32 + kb);
            acc = __builtin_amdgcn_wmma_f32_16x16x32_bf16(false, a, false, b,
                                                          (short)0, acc, false, false);
        }
        const float bias = bqkv[2 * C + n0 + lh];
        unsigned short* vrow = sVT + (n0 + lh) * RS_VT;    // transposed row = v column
        for (int r = 0; r < 8; r += 2) {
            unsigned int p = f2bf2(acc[r] + bias, acc[r + 1] + bias);
            *(unsigned int*)&vrow[m0 + r + kb] = p;        // (m even) -> one b32 store
        }
    }
    __syncthreads();

    // ---- Phase C: scores = (q*scale) @ k^T  (4 heads * 4x4 tiles, K=32) ----
    for (int t0 = wave; t0 < 64; t0 += 8) {
        const int t  = __builtin_amdgcn_readfirstlane(t0);
        const int h  = t >> 4, r2 = t & 15;
        const int m0 = (r2 >> 2) << 4, n0 = (r2 & 3) << 4;
        v16bf a = ldfrag(sQK + (m0 + lh) * RS_QK + h * HD + kb);       // q rows
        v16bf b = ldfrag(sQK + (n0 + lh) * RS_QK + C + h * HD + kb);   // k rows = B cols
        f32x8 acc = {};
        acc = __builtin_amdgcn_wmma_f32_16x16x32_bf16(false, a, false, b,
                                                      (short)0, acc, false, false);
        for (int r = 0; r < 8; ++r) {
            int m = m0 + r + kb;
            sS[(h * TP + m) * RS_S + n0 + lh] = acc[r];
        }
    }
    __syncthreads();

    // ---- Phase D: row softmax in registers; bf16 probs in place ----
    {
        int h = tid >> 6, m = tid & 63;                 // 4*64 rows == 256 threads
        float* row = &sS[(h * TP + m) * RS_S];
        unsigned short* prow = (unsigned short*)row;
        uint2 z2; z2.x = 0u; z2.y = 0u;
        if (m < T) {
            const float* mrow = maskP + ((size_t)((w & (NWPI - 1)) * TP + m)) * RS_S;
            float4 buf[13];                              // k = 0..51 (>=49 forced ~ -1e30)
            float mx = -1e30f;
            for (int j = 0; j < 13; ++j) {
                float4 s  = *(const float4*)(row + 4 * j);
                float4 mk = *(const float4*)(mrow + 4 * j);
                s.x += mk.x; s.y += mk.y; s.z += mk.z; s.w += mk.w;
                buf[j] = s;
                mx = fmaxf(mx, fmaxf(fmaxf(s.x, s.y), fmaxf(s.z, s.w)));
            }
            float sum = 0.0f;
            for (int j = 0; j < 13; ++j) {
                float4 s = buf[j];
                s.x = __expf(s.x - mx); s.y = __expf(s.y - mx);
                s.z = __expf(s.z - mx); s.w = __expf(s.w - mx);
                sum += (s.x + s.y) + (s.z + s.w);
                buf[j] = s;
            }
            float inv = 1.0f / sum;
            for (int j = 0; j < 13; ++j) {               // packed bf16, 4/store
                uint2 p;
                p.x = f2bf2(buf[j].x * inv, buf[j].y * inv);
                p.y = f2bf2(buf[j].z * inv, buf[j].w * inv);
                *(uint2*)(prow + 4 * j) = p;             // writes trail f32 reads
            }
            *(uint2*)(prow + 52) = z2;                   // zero pad cols 52..63
            *(uint2*)(prow + 56) = z2;
            *(uint2*)(prow + 60) = z2;
        } else {
            for (int j = 0; j < 16; ++j) *(uint2*)(prow + 4 * j) = z2;  // zero pad rows
        }
    }
    __syncthreads();

    // ---- Phase E: o = probs @ v  (4 heads * 4x2 tiles, K=64) into sX ----
    for (int t0 = wave; t0 < 32; t0 += 8) {
        const int t  = __builtin_amdgcn_readfirstlane(t0);
        const int h  = t >> 3, r2 = t & 7;
        const int m0 = (r2 >> 1) << 4, d0 = (r2 & 1) << 4;
        f32x8 acc = {};
        const unsigned short* prow = (const unsigned short*)&sS[(h * TP + m0 + lh) * RS_S];
        const unsigned short* vrow = sVT + (h * HD + d0 + lh) * RS_VT;
        for (int ks = 0; ks < 2; ++ks) {
            v16bf a = ldfrag(prow + ks * 32 + kb);
            v16bf b = ldfrag(vrow + ks * 32 + kb);
            acc = __builtin_amdgcn_wmma_f32_16x16x32_bf16(false, a, false, b,
                                                          (short)0, acc, false, false);
        }
        const int col = h * HD + d0 + lh;                  // [t][h*32+d] == [t][c]
        for (int r = 0; r < 8; r += 2) {
            unsigned int p = f2bf2(acc[r], acc[r + 1]);
            int m = m0 + r + kb;
            sX[m * RS_X + col]       = (unsigned short)p;
            sX[(m + 1) * RS_X + col] = (unsigned short)(p >> 16);
        }
    }
    __syncthreads();

    // ---- Phase F: out = o @ Wproj + b  (4x8 tiles, K=128), f32 store ----
    float* ow = out + (size_t)w * T * C;
    for (int t0 = wave; t0 < 32; t0 += 8) {
        const int t  = __builtin_amdgcn_readfirstlane(t0);
        const int m0 = (t >> 3) << 4, n0 = (t & 7) << 4;
        f32x8 acc = {};
        const unsigned short* arow = sX + (m0 + lh) * RS_X;
        const unsigned short* brow = WprojT + (n0 + lh) * C;
        for (int kk = 0; kk < 4; ++kk) {
            v16bf a = ldfrag(arow + kk * 32 + kb);
            v16bf b = ldfrag(brow + kk * 32 + kb);
            acc = __builtin_amdgcn_wmma_f32_16x16x32_bf16(false, a, false, b,
                                                          (short)0, acc, false, false);
        }
        const float bias = bproj[n0 + lh];
        for (int r = 0; r < 8; ++r) {
            int m = m0 + r + kb;
            if (m < T) ow[m * C + n0 + lh] = acc[r] + bias;
        }
    }
}

extern "C" void kernel_launch(void* const* d_in, const int* in_sizes, int n_in,
                              void* d_out, int out_size, void* d_ws, size_t ws_size,
                              hipStream_t stream) {
    (void)in_sizes; (void)n_in; (void)out_size; (void)ws_size;
    const float* x     = (const float*)d_in[0];
    const float* mask  = (const float*)d_in[1];
    const float* Wqkv  = (const float*)d_in[2];
    const float* bqkv  = (const float*)d_in[3];
    const float* Wproj = (const float*)d_in[4];
    const float* bproj = (const float*)d_in[5];

    unsigned short* WqkvT  = (unsigned short*)d_ws;            // 384*128 bf16
    unsigned short* WprojT = WqkvT + 3 * C * C;                // 128*128 bf16
    float*          maskP  = (float*)(WprojT + C * C);         // [64][64][68] f32

    wattn_prep_weights<<<(3 * C * C + C * C) / 256, 256, 0, stream>>>(Wqkv, Wproj, WqkvT, WprojT);
    wattn_prep_mask<<<(NWPI * TP * RS_S + 255) / 256, 256, 0, stream>>>(mask, maskP);
    wattn_main<<<NW_TOTAL, 256, 0, stream>>>(x, maskP, bqkv, bproj, WqkvT, WprojT, (float*)d_out);
}
